// BeamSearchDecoder_47545287967568
// MI455X (gfx1250) — matrix operand order repaired and checked by reading, not compiled, not run on hardware
//
#include <hip/hip_runtime.h>
#include <math.h>

#define SLEN 512
#define BATCH 32
#define KBEAM 8
#define BK 256      // BATCH*KBEAM
#define DKV 256
#define HID 512
#define EMBD 256
#define VOC 34
#define VPAD 48
#define ML 64
#define PAD_T 0
#define SOS_T 1
#define EOS_T 2

typedef __attribute__((ext_vector_type(16))) __bf16 v16bf;
typedef __attribute__((ext_vector_type(8)))  float  v8f;

#if __has_builtin(__builtin_amdgcn_tensor_load_to_lds) && __has_builtin(__builtin_amdgcn_s_wait_tensorcnt)
#define HAVE_TDM 1
#else
#define HAVE_TDM 0
#endif

// LDS tile pitch: 32 bf16 payload (64B) + 2 bf16 pad (4B) per row
#define TPITCH 34

// ---------------- WMMA fragment helpers (CDNA5 wave32 layouts) ----------------
// A-matrix 16x32 bf16: lane L -> m = L&15, half = L>>4.
// elems 0..7 are A[m][half*8 + 0..7], elems 8..15 are A[m][16 + half*8 + 0..7]
__device__ __forceinline__ v16bf frag_a16(const __bf16* base, int lane, int pitch) {
  const int half = (lane >> 4) & 1, m = lane & 15;
  const __bf16* p = base + m * pitch + half * 8;
  v16bf a;
#pragma unroll
  for (int e = 0; e < 8; ++e) { a[e] = p[e]; a[8 + e] = p[16 + e]; }
  return a;
}

// B-matrix 32x16 bf16 stored K-transposed in LDS: lBt[n][k].
// lane L -> n = nloc + (L&15), half = L>>4; elems 0..7 = B[half*8 + 0..7][n], elems 8..15 = B[16 + half*8 + 0..7][n]
__device__ __forceinline__ v16bf frag_b16(const __bf16* lBt, int lane, int pitch, int nloc) {
  const int half = (lane >> 4) & 1, n = nloc + (lane & 15);
  const __bf16* p = lBt + n * pitch + half * 8;
  v16bf b;
#pragma unroll
  for (int e = 0; e < 8; ++e) { b[e] = p[e]; b[8 + e] = p[16 + e]; }
  return b;
}

// ---------------- TDM tile loader: [rows x 32 bf16], row stride in elems ------
#if HAVE_TDM
typedef __attribute__((ext_vector_type(4))) unsigned int v4u_t;
typedef __attribute__((ext_vector_type(8))) int v8i_t;
typedef __attribute__((ext_vector_type(4))) int v4i_t;

__device__ __forceinline__ void tdm_tile(const __bf16* gbase, unsigned rowStride,
                                         int rows, __bf16* ldst) {
  const unsigned long long ga = (unsigned long long)(size_t)gbase;
  const unsigned ldsa = (unsigned)(size_t)ldst;   // low 32 bits of flat ptr = LDS offset
  v4u_t g0;
  g0[0] = 1u;                                              // count=1, user descriptor
  g0[1] = ldsa;                                            // lds_addr (bytes)
  g0[2] = (unsigned)(ga & 0xffffffffull);                  // global_addr[31:0]
  g0[3] = (unsigned)((ga >> 32) & 0x01ffffffull) | (2u << 30); // addr[56:32], type=2
  v8i_t g1;
  g1[0] = (int)((1u << 16)        // data_size = 1 -> 2 bytes (bf16)
              | (1u << 20)        // pad_enable
              | (3u << 22));      // pad_interval = 16 DWORDs (one 64B row)
                                  // pad_amount = 0 -> 1 DWORD pad -> pitch 34 bf16
  g1[1] = (int)(32u << 16);                 // tensor_dim0[15:0] = 32 (in [31:16])
  g1[2] = (int)(((unsigned)rows) << 16);    // tensor_dim1[15:0] = rows (in [31:16])
  g1[3] = (int)(32u << 16);                 // tile_dim0 = 32 (in [31:16])
  g1[4] = (int)((unsigned)rows & 0xffffu);  // tile_dim1 = rows; tile_dim2 = 0
  g1[5] = (int)rowStride;                   // tensor_dim0_stride[31:0]
  g1[6] = 0;
  g1[7] = 0;
  v4i_t z4 = {0, 0, 0, 0};
#if __clang_major__ >= 23
  v8i_t z8 = {0, 0, 0, 0, 0, 0, 0, 0};
  __builtin_amdgcn_tensor_load_to_lds(g0, g1, z4, z4, z8, 0);
#else
  __builtin_amdgcn_tensor_load_to_lds(g0, g1, z4, z4, 0);
#endif
}
#endif

// Stage a [rows x 32] bf16 tile (row stride rowStride elems) into lT (pitch 34).
__device__ __forceinline__ void load_tile_tr(const __bf16* gbase, int rowStride,
                                             int rows, __bf16* lT, int tid) {
#if HAVE_TDM
  if ((tid >> 5) == 0) tdm_tile(gbase, (unsigned)rowStride, rows, lT);
  __builtin_amdgcn_s_wait_tensorcnt(0);
#else
#pragma unroll
  for (int r = 0; r < 16; ++r) {
    const int idx = r * 256 + tid;
    const int nn = idx >> 5, kk = idx & 31;
    lT[nn * TPITCH + kk] = (nn < rows) ? gbase[(size_t)nn * rowStride + kk] : (__bf16)0.f;
  }
#endif
}

// ---------------- generic bf16 GEMM: C[256 x N] = A[256 x K] * Bt[N x K] ------
// Bt is the transposed (N-major) weight matrix; block: 8 waves, tile 16(M)x128(N)
__global__ __launch_bounds__(256) void gemm_bf16_k(
    const __bf16* __restrict__ A, int lda,
    const __bf16* __restrict__ Bt,
    float* __restrict__ C, int ldc, int N, int K)
{
  __shared__ __bf16 lA[16 * 32];
  __shared__ __bf16 lBt[128 * TPITCH];
  const int tid = threadIdx.x, wid = tid >> 5, lane = tid & 31;
  const int mbase = blockIdx.y * 16, n0 = blockIdx.x * 128;
  const int rows = (N - n0 < 128) ? (N - n0) : 128;
  v8f acc = {};
  for (int k0 = 0; k0 < K; k0 += 32) {
    {
      const int e = tid * 2, r = e >> 5, c = e & 31;
      const __bf16* ap = A + (mbase + r) * lda + k0 + c;
      lA[e] = ap[0];
      lA[e + 1] = ap[1];
    }
    if (k0 + 32 < K)
      __builtin_prefetch(Bt + (size_t)(n0 + (tid & 127)) * K + k0 + 32, 0, 1);
    load_tile_tr(Bt + (size_t)n0 * K + k0, K, rows, lBt, tid);
    __syncthreads();
    if (wid * 16 < rows) {
      v16bf a = frag_a16(lA, lane, 32);
      v16bf b = frag_b16(lBt, lane, TPITCH, wid * 16);
      acc = __builtin_amdgcn_wmma_f32_16x16x32_bf16(false, a, false, b, (short)0, acc, false, false);
    }
    __syncthreads();
  }
  const int ntile = n0 + wid * 16;
  if (ntile < N) {
    const int half = lane >> 4, n = lane & 15;
#pragma unroll
    for (int v = 0; v < 8; ++v)
      C[(mbase + v + 8 * half) * ldc + ntile + n] = acc[v];
  }
}

// ---------------- attention: one block per batch entry -----------------------
// keysbf: [S][B][DK]  (rows along s contiguous in d -> TDM friendly)
// Vt:     [B][DV][S]  (rows along d contiguous in s -> TDM friendly)
__global__ __launch_bounds__(256) void attention_k(
    const __bf16* __restrict__ qbf, const __bf16* __restrict__ keysbf,
    const __bf16* __restrict__ Vt, const int* __restrict__ src_len,
    __bf16* __restrict__ ctxbf, float* __restrict__ attn_out)
{
  __shared__ __attribute__((aligned(16))) unsigned char smem[49664];
  float*  lE    = (float*)smem;                 // 8  x 512 f32   (16384 B)
  __bf16* lattn = (__bf16*)(smem + 16384);      // 16 x 512 bf16  (16384 B)
  __bf16* lq    = (__bf16*)(smem + 32768);      // 16 x 256 bf16  (8192 B)
  __bf16* lT    = (__bf16*)(smem + 40960);      // 128 x 34 bf16  (8704 B)
  const int b = blockIdx.x;
  const int tid = threadIdx.x, wid = tid >> 5, lane = tid & 31;

  // load q rows (pad rows 8..15 with zeros)
#pragma unroll
  for (int r = 0; r < 16; ++r) {
    const int idx = r * 256 + tid;
    const int m = idx >> 8, d = idx & 255;
    lq[idx] = (m < KBEAM) ? qbf[(b * KBEAM + m) * DKV + d] : (__bf16)0.f;
  }
  __syncthreads();

  // energy[8 x 512] = q[8 x 256] @ keys_b^T
  for (int npass = 0; npass < 4; ++npass) {
    v8f acc = {};
    for (int d0 = 0; d0 < DKV; d0 += 32) {
      load_tile_tr(keysbf + (size_t)(npass * 128) * (BATCH * DKV) + b * DKV + d0,
                   BATCH * DKV, 128, lT, tid);
      __syncthreads();
      v16bf a  = frag_a16(lq + d0, lane, DKV);
      v16bf bb = frag_b16(lT, lane, TPITCH, wid * 16);
      acc = __builtin_amdgcn_wmma_f32_16x16x32_bf16(false, a, false, bb, (short)0, acc, false, false);
      __syncthreads();
    }
    if ((lane >> 4) == 0) {       // only rows m = 0..7 are valid
      const int n = npass * 128 + wid * 16 + (lane & 15);
#pragma unroll
      for (int v = 0; v < 8; ++v) lE[v * SLEN + n] = acc[v];
    }
    __syncthreads();
  }

  // masked softmax: wave w handles row w
  {
    const int row = wid;
    const int sl = src_len[b];
    float vals[16];
    float mx = -__builtin_inff();
#pragma unroll
    for (int j = 0; j < 16; ++j) {
      const int s = lane + 32 * j;
      float e = lE[row * SLEN + s];
      e = (s < sl) ? e : -1e9f;
      vals[j] = e;
      mx = fmaxf(mx, e);
    }
#pragma unroll
    for (int m = 16; m >= 1; m >>= 1) mx = fmaxf(mx, __shfl_xor(mx, m, 32));
    float sum = 0.f;
#pragma unroll
    for (int j = 0; j < 16; ++j) { vals[j] = __expf(vals[j] - mx); sum += vals[j]; }
#pragma unroll
    for (int m = 16; m >= 1; m >>= 1) sum += __shfl_xor(sum, m, 32);
    const float inv = 1.f / sum;
#pragma unroll
    for (int j = 0; j < 16; ++j) {
      const int s = lane + 32 * j;
      const float at = vals[j] * inv;
      attn_out[(size_t)(b * KBEAM + row) * SLEN + s] = at;
      lattn[row * SLEN + s] = (__bf16)at;
    }
  }
  // zero pad rows 8..15 of attn
#pragma unroll
  for (int r = 0; r < 16; ++r) {
    const int idx = r * 256 + tid;
    if (idx < 8 * SLEN) lattn[8 * SLEN + idx] = (__bf16)0.f;
  }
  __syncthreads();

  // ctx[8 x 256] = attn[8 x 512] @ values_b[512 x 256]
  for (int npass = 0; npass < 2; ++npass) {
    v8f acc = {};
    for (int s0 = 0; s0 < SLEN; s0 += 32) {
      load_tile_tr(Vt + ((size_t)b * DKV + npass * 128) * SLEN + s0, SLEN, 128, lT, tid);
      __syncthreads();
      v16bf a  = frag_a16(lattn + s0, lane, SLEN);
      v16bf bb = frag_b16(lT, lane, TPITCH, wid * 16);
      acc = __builtin_amdgcn_wmma_f32_16x16x32_bf16(false, a, false, bb, (short)0, acc, false, false);
      __syncthreads();
    }
    if ((lane >> 4) == 0) {
      const int n = npass * 128 + wid * 16 + (lane & 15);
#pragma unroll
      for (int v = 0; v < 8; ++v)
        ctxbf[(size_t)(b * KBEAM + v) * DKV + n] = (__bf16)acc[v];
    }
    __syncthreads();
  }
}

// ---------------- elementwise / glue kernels ---------------------------------
__device__ __forceinline__ float sigf(float x) { return 1.f / (1.f + __expf(-x)); }

__global__ __launch_bounds__(256) void lstm_elem_k(
    const float* __restrict__ gates, const float* __restrict__ bias,
    float* __restrict__ c, __bf16* __restrict__ h)
{
  const int idx = blockIdx.x * 256 + threadIdx.x;   // over 256*512
  const int row = idx >> 9, j = idx & 511;
  const float* g = gates + row * 2048;
  const float gi = g[j] + bias[j];
  const float gf = g[512 + j] + bias[512 + j];
  const float gg = g[1024 + j] + bias[1024 + j];
  const float go = g[1536 + j] + bias[1536 + j];
  const float cn = sigf(gf) * c[idx] + sigf(gi) * tanhf(gg);
  c[idx] = cn;
  h[row * 1024 + j] = (__bf16)(sigf(go) * tanhf(cn));
}

__global__ __launch_bounds__(256) void build_a1_k(
    const __bf16* __restrict__ Ebf, const int* __restrict__ tokens,
    const __bf16* __restrict__ ctxbf, const __bf16* __restrict__ HB,
    __bf16* __restrict__ A1)
{
  const int row = blockIdx.x, tid = threadIdx.x;
  const int tok = tokens[row];
#pragma unroll
  for (int t = 0; t < 4; ++t) {
    const int col = tid + t * 256;
    __bf16 v;
    if (col < 256)      v = Ebf[tok * EMBD + col];
    else if (col < 512) v = ctxbf[row * DKV + (col - 256)];
    else                v = HB[row * 1024 + (col - 512)];   // h1
    A1[row * 1024 + col] = v;
  }
}

__global__ __launch_bounds__(256) void build_a4_k(
    const __bf16* __restrict__ HB, const __bf16* __restrict__ ctxbf,
    __bf16* __restrict__ A4)
{
  const int row = blockIdx.x, tid = threadIdx.x;
#pragma unroll
  for (int t = 0; t < 3; ++t) {
    const int col = tid + t * 256;
    const __bf16 v = (col < 512) ? HB[row * 1024 + 512 + col]        // h2
                                 : ctxbf[row * DKV + (col - 512)];
    A4[row * 768 + col] = v;
  }
}

__global__ void cvt_f32_bf16_k(const float* __restrict__ s, __bf16* __restrict__ d, int n) {
  const int i = blockIdx.x * blockDim.x + threadIdx.x;
  if (i < n) d[i] = (__bf16)s[i];
}
__global__ void zero_bf16_k(__bf16* p, int n) {
  const int i = blockIdx.x * blockDim.x + threadIdx.x;
  if (i < n) p[i] = (__bf16)0.f;
}
__global__ void zero_f32_k(float* p, int n) {
  const int i = blockIdx.x * blockDim.x + threadIdx.x;
  if (i < n) p[i] = 0.f;
}

// pack [Wx;Wh] (each 512 x 2048, K-major) -> transposed bf16 [2048][1024]
__global__ void pack_w12t_k(const float* __restrict__ wx, const float* __restrict__ wh,
                            __bf16* __restrict__ d) {
  const int i = blockIdx.x * 256 + threadIdx.x;   // over 2048*1024
  if (i < 2048 * 1024) {
    const int k = i & 1023, n = i >> 10;
    const float v = (k < 512) ? wx[k * 2048 + n] : wh[(k - 512) * 2048 + n];
    d[i] = (__bf16)v;
  }
}
// Wq (512 x 256) -> [256][512]
__global__ void pack_wqt_k(const float* __restrict__ w, __bf16* __restrict__ d) {
  const int i = blockIdx.x * 256 + threadIdx.x;   // over 256*512
  if (i < 256 * 512) {
    const int k = i & 511, n = i >> 9;
    d[i] = (__bf16)w[k * 256 + n];
  }
}
// Wout (768 x 34) -> [48][768], zero-padded rows n>=34
__global__ void pack_wott_k(const float* __restrict__ w, __bf16* __restrict__ d) {
  const int i = blockIdx.x * 256 + threadIdx.x;   // over 48*768
  if (i < VPAD * 768) {
    const int k = i % 768, n = i / 768;
    d[i] = (n < VOC) ? (__bf16)w[k * VOC + n] : (__bf16)0.f;
  }
}
// values [S][B][DV] f32 -> Vt [B][DV][S] bf16
__global__ void pack_vals_t_k(const float* __restrict__ v, __bf16* __restrict__ d) {
  const int i = blockIdx.x * 256 + threadIdx.x;   // over B*DV*S
  if (i < BATCH * DKV * SLEN) {
    const int s = i & 511;
    const int t = i >> 9;
    const int dd = t & 255, b = t >> 8;
    d[i] = (__bf16)v[(size_t)s * (BATCH * DKV) + b * DKV + dd];
  }
}
__global__ void init_seq_k(int* seq, int* tokens) {
  const int i = blockIdx.x * 256 + threadIdx.x;
  if (i < ML * BATCH * KBEAM) seq[i] = (i < BATCH * KBEAM) ? SOS_T : PAD_T;
  if (i < BK) tokens[i] = SOS_T;
}

// ---------------- beam pruning ------------------------------------------------
__global__ __launch_bounds__(32) void prune0_k(
    const float* __restrict__ logits, const float* __restrict__ bout,
    float* scores, int* active, int* lengths, int* seq, int* tokens)
{
  const int b = blockIdx.x, lane = threadIdx.x;
  __shared__ float lp[VOC];
  __shared__ float cs[KBEAM]; __shared__ int ct[KBEAM];
  __shared__ int act_s[KBEAM];
  if (lane == 0) {
    const float* lg = logits + (b * KBEAM) * VPAD;  // rows within batch identical
    float mx = -__builtin_inff();
    for (int v = 0; v < VOC; ++v) mx = fmaxf(mx, lg[v] + bout[v]);
    float s = 0.f;
    for (int v = 0; v < VOC; ++v) s += __expf(lg[v] + bout[v] - mx);
    const float lse = mx + logf(s);
    for (int v = 0; v < VOC; ++v) lp[v] = lg[v] + bout[v] - lse;
  }
  __syncthreads();
  float cand[2]; int cidx[2];
#pragma unroll
  for (int j = 0; j < 2; ++j) {
    const int idx = lane + 32 * j;
    cidx[j] = idx;
    cand[j] = (idx < VOC) ? lp[idx] : -__builtin_inff();
  }
  for (int sel = 0; sel < KBEAM; ++sel) {
    float bv = -__builtin_inff(); int bi = 1 << 30;
#pragma unroll
    for (int j = 0; j < 2; ++j)
      if (cand[j] > bv || (cand[j] == bv && cidx[j] < bi)) { bv = cand[j]; bi = cidx[j]; }
#pragma unroll
    for (int m = 16; m >= 1; m >>= 1) {
      const float ov = __shfl_xor(bv, m, 32); const int oi = __shfl_xor(bi, m, 32);
      if (ov > bv || (ov == bv && oi < bi)) { bv = ov; bi = oi; }
    }
    if (lane == 0) { cs[sel] = bv; ct[sel] = bi; }
#pragma unroll
    for (int j = 0; j < 2; ++j) if (cidx[j] == bi) cand[j] = -__builtin_inff();
  }
  __syncthreads();
  if (lane < KBEAM) {
    const int tok = ct[lane];
    scores[b * KBEAM + lane] = cs[lane];
    seq[(0 * BATCH + b) * KBEAM + lane] = SOS_T;
    seq[(1 * BATCH + b) * KBEAM + lane] = tok;
    tokens[b * KBEAM + lane] = tok;
    const int a = (tok != EOS_T);
    act_s[lane] = a;
    lengths[b * KBEAM + lane] = a ? 0 : 2;
  }
  __syncthreads();
  if (lane < KBEAM) {
    int top = 0; float bv = cs[0];
    for (int k = 1; k < KBEAM; ++k) if (cs[k] > bv) { bv = cs[k]; top = k; }
    active[b * KBEAM + lane] = act_s[lane] && act_s[top];
  }
}

__global__ __launch_bounds__(32) void prune_step_k(
    const float* __restrict__ logits, const float* __restrict__ bout,
    float* scores, int* active, int* lengths, int* seq, int* tokens, int step_i)
{
  const int b = blockIdx.x, lane = threadIdx.x;
  __shared__ float lp[KBEAM * VOC];
  __shared__ float sc_s[KBEAM]; __shared__ int act_s[KBEAM];
  __shared__ float cs[KBEAM]; __shared__ int crow[KBEAM], ctok[KBEAM];
  __shared__ float nsc_s[KBEAM]; __shared__ int src_s[KBEAM], ntok_s[KBEAM];
  __shared__ int oldseq[ML * KBEAM];

  if (lane < KBEAM) {
    const float* lg = logits + (b * KBEAM + lane) * VPAD;
    float mx = -__builtin_inff();
    for (int v = 0; v < VOC; ++v) mx = fmaxf(mx, lg[v] + bout[v]);
    float s = 0.f;
    for (int v = 0; v < VOC; ++v) s += __expf(lg[v] + bout[v] - mx);
    const float lse = mx + logf(s);
    for (int v = 0; v < VOC; ++v) lp[lane * VOC + v] = lg[v] + bout[v] - lse;
    sc_s[lane] = scores[b * KBEAM + lane];
    act_s[lane] = active[b * KBEAM + lane];
  }
  __syncthreads();

  // top-8 (sorted, index tie-break ascending) over 8*34 = 272 candidates
  float cand[9]; int cidx[9];
#pragma unroll
  for (int j = 0; j < 9; ++j) {
    const int idx = lane * 9 + j;
    cidx[j] = idx;
    if (idx < KBEAM * VOC) {
      const int k = idx / VOC;
      cand[j] = act_s[k] ? (sc_s[k] + lp[idx]) : -__builtin_inff();
    } else cand[j] = -__builtin_inff();
  }
  for (int sel = 0; sel < KBEAM; ++sel) {
    float bv = -__builtin_inff(); int bi = 1 << 30;
#pragma unroll
    for (int j = 0; j < 9; ++j)
      if (cand[j] > bv || (cand[j] == bv && cidx[j] < bi)) { bv = cand[j]; bi = cidx[j]; }
#pragma unroll
    for (int m = 16; m >= 1; m >>= 1) {
      const float ov = __shfl_xor(bv, m, 32); const int oi = __shfl_xor(bi, m, 32);
      if (ov > bv || (ov == bv && oi < bi)) { bv = ov; bi = oi; }
    }
    if (lane == 0) { cs[sel] = bv; crow[sel] = bi / VOC; ctok[sel] = bi % VOC; }
#pragma unroll
    for (int j = 0; j < 9; ++j) if (cidx[j] == bi) cand[j] = -__builtin_inff();
  }
  __syncthreads();

  if (lane < KBEAM) {
    int rank = -1;
    for (int kk = 0; kk <= lane; ++kk) rank += act_s[kk];
    const bool a = act_s[lane] != 0;
    nsc_s[lane]  = a ? cs[rank]   : sc_s[lane];
    src_s[lane]  = a ? crow[rank] : lane;
    ntok_s[lane] = a ? ctok[rank] : PAD_T;
  }
  __syncthreads();

  // reorder seq along beam dim (stage through LDS)
#pragma unroll
  for (int r = 0; r < 16; ++r) {
    const int idx = r * 32 + lane;
    const int t = idx >> 3, k = idx & 7;
    oldseq[idx] = seq[(t * BATCH + b) * KBEAM + src_s[k]];
  }
  __syncthreads();
#pragma unroll
  for (int r = 0; r < 16; ++r) {
    const int idx = r * 32 + lane;
    const int t = idx >> 3, k = idx & 7;
    seq[(t * BATCH + b) * KBEAM + k] = oldseq[idx];
  }
  __syncthreads();

  if (lane < KBEAM) {
    const int ntk = ntok_s[lane];
    seq[((step_i + 1) * BATCH + b) * KBEAM + lane] = ntk;
    tokens[b * KBEAM + lane] = ntk;
    const bool eos = (ntk == EOS_T);
    if (eos) lengths[b * KBEAM + lane] = step_i + 2;
    act_s[lane] = (act_s[lane] != 0) && !eos;
    scores[b * KBEAM + lane] = nsc_s[lane];
  }
  __syncthreads();
  if (lane < KBEAM) {
    int top = 0; float bv = nsc_s[0];
    for (int k = 1; k < KBEAM; ++k) if (nsc_s[k] > bv) { bv = nsc_s[k]; top = k; }
    active[b * KBEAM + lane] = act_s[lane] && act_s[top];
  }
}

__global__ __launch_bounds__(64) void finalize_k(
    const int* __restrict__ seq, const float* __restrict__ scores,
    const int* __restrict__ lengths, float* __restrict__ out)
{
  const int b = blockIdx.x, t = threadIdx.x;
  __shared__ int top_s;
  if (t == 0) {
    int top = 0; float bv = scores[b * KBEAM];
    for (int k = 1; k < KBEAM; ++k) {
      const float v = scores[b * KBEAM + k];
      if (v > bv) { bv = v; top = k; }
    }
    top_s = top;
  }
  __syncthreads();
  const int tok = (t == ML - 1) ? EOS_T : seq[(t * BATCH + b) * KBEAM + top_s];
  out[t * BATCH + b] = (float)tok;
  if (t == 0) out[ML * BATCH + b] = (float)lengths[b * KBEAM + top_s];
}

// ---------------- host orchestration -----------------------------------------
extern "C" void kernel_launch(void* const* d_in, const int* in_sizes, int n_in,
                              void* d_out, int out_size, void* d_ws, size_t ws_size,
                              hipStream_t stream)
{
  (void)in_sizes; (void)n_in; (void)out_size; (void)ws_size;
  const float* keys   = (const float*)d_in[0];
  const float* values = (const float*)d_in[1];
  const int*   srcl   = (const int*)d_in[2];
  const float* E    = (const float*)d_in[5];
  const float* Wx1  = (const float*)d_in[6];
  const float* Wh1  = (const float*)d_in[7];
  const float* b1   = (const float*)d_in[8];
  const float* Wx2  = (const float*)d_in[9];
  const float* Wh2  = (const float*)d_in[10];
  const float* b2   = (const float*)d_in[11];
  const float* Wq   = (const float*)d_in[12];
  const float* Wout = (const float*)d_in[13];
  const float* bout = (const float*)d_in[14];

  char* ws = (char*)d_ws;
  size_t off = 0;
  auto carve = [&](size_t bytes) -> char* {
    off = (off + 255) & ~(size_t)255;
    char* p = ws + off;
    off += bytes;
    return p;
  };
  __bf16* W1t   = (__bf16*)carve((size_t)2048 * 1024 * 2);   // [N][K]
  __bf16* W2t   = (__bf16*)carve((size_t)2048 * 1024 * 2);
  __bf16* Wqt   = (__bf16*)carve((size_t)256 * 512 * 2);
  __bf16* Wot   = (__bf16*)carve((size_t)VPAD * 768 * 2);
  __bf16* Ebf   = (__bf16*)carve((size_t)VOC * EMBD * 2);
  __bf16* Kbf   = (__bf16*)carve((size_t)SLEN * BATCH * DKV * 2);   // [S][B][DK]
  __bf16* Vt    = (__bf16*)carve((size_t)BATCH * DKV * SLEN * 2);   // [B][DV][S]
  __bf16* A1    = (__bf16*)carve((size_t)BK * 1024 * 2);
  __bf16* HB    = (__bf16*)carve((size_t)BK * 1024 * 2);   // [h1 | h2]
  __bf16* A4    = (__bf16*)carve((size_t)BK * 768 * 2);
  __bf16* ctxbf = (__bf16*)carve((size_t)BK * DKV * 2);
  __bf16* qbf   = (__bf16*)carve((size_t)BK * DKV * 2);
  float*  qf    = (float*)carve((size_t)BK * DKV * 4);
  float*  gates = (float*)carve((size_t)BK * 2048 * 4);
  float*  logit = (float*)carve((size_t)BK * VPAD * 4);
  float*  c1    = (float*)carve((size_t)BK * HID * 4);
  float*  c2    = (float*)carve((size_t)BK * HID * 4);
  float*  scores= (float*)carve((size_t)BATCH * KBEAM * 4);
  int*    active= (int*)carve((size_t)BATCH * KBEAM * 4);
  int*    lens  = (int*)carve((size_t)BATCH * KBEAM * 4);
  int*    seq   = (int*)carve((size_t)ML * BATCH * KBEAM * 4);
  int*    tokens= (int*)carve((size_t)BK * 4);

  float* out   = (float*)d_out;
  float* attns = out + ML * BATCH + BATCH;   // padded(2048) + lens(32)

  // one-time (per call) bf16 packing (weights transposed to N-major for TDM)
  pack_w12t_k<<<(2048 * 1024 + 255) / 256, 256, 0, stream>>>(Wx1, Wh1, W1t);
  pack_w12t_k<<<(2048 * 1024 + 255) / 256, 256, 0, stream>>>(Wx2, Wh2, W2t);
  pack_wqt_k<<<(256 * 512 + 255) / 256, 256, 0, stream>>>(Wq, Wqt);
  pack_wott_k<<<(VPAD * 768 + 255) / 256, 256, 0, stream>>>(Wout, Wot);
  cvt_f32_bf16_k<<<(VOC * EMBD + 255) / 256, 256, 0, stream>>>(E, Ebf, VOC * EMBD);
  const int NKV = SLEN * BATCH * DKV;
  cvt_f32_bf16_k<<<(NKV + 255) / 256, 256, 0, stream>>>(keys, Kbf, NKV);
  pack_vals_t_k<<<(NKV + 255) / 256, 256, 0, stream>>>(values, Vt);

  // state init
  zero_bf16_k<<<(BK * 1024 + 255) / 256, 256, 0, stream>>>(HB, BK * 1024);
  zero_bf16_k<<<(BK * DKV + 255) / 256, 256, 0, stream>>>(ctxbf, BK * DKV);
  zero_f32_k<<<(BK * HID + 255) / 256, 256, 0, stream>>>(c1, BK * HID);
  zero_f32_k<<<(BK * HID + 255) / 256, 256, 0, stream>>>(c2, BK * HID);
  init_seq_k<<<(ML * BATCH * KBEAM + 255) / 256, 256, 0, stream>>>(seq, tokens);

  // 63 decoder steps: t=0 is the SOS expansion, t=1..62 are the scan body
  for (int t = 0; t < ML - 1; ++t) {
    build_a1_k<<<BK, 256, 0, stream>>>(Ebf, tokens, ctxbf, HB, A1);
    gemm_bf16_k<<<dim3(16, 16), 256, 0, stream>>>(A1, 1024, W1t, gates, 2048, 2048, 1024);
    lstm_elem_k<<<(BK * HID) / 256, 256, 0, stream>>>(gates, b1, c1, HB);          // h1
    gemm_bf16_k<<<dim3(16, 16), 256, 0, stream>>>(HB, 1024, W2t, gates, 2048, 2048, 1024);
    lstm_elem_k<<<(BK * HID) / 256, 256, 0, stream>>>(gates, b2, c2, HB + 512);    // h2
    gemm_bf16_k<<<dim3(2, 16), 256, 0, stream>>>(HB + 512, 1024, Wqt, qf, DKV, DKV, HID);
    cvt_f32_bf16_k<<<(BK * DKV + 255) / 256, 256, 0, stream>>>(qf, qbf, BK * DKV);
    attention_k<<<BATCH, 256, 0, stream>>>(qbf, Kbf, Vt, srcl, ctxbf,
                                           attns + (size_t)t * BK * SLEN);
    build_a4_k<<<BK, 256, 0, stream>>>(HB, ctxbf, A4);
    gemm_bf16_k<<<dim3(1, 16), 256, 0, stream>>>(A4, 768, Wot, logit, VPAD, VPAD, 768);
    if (t == 0)
      prune0_k<<<BATCH, 32, 0, stream>>>(logit, bout, scores, active, lens, seq, tokens);
    else
      prune_step_k<<<BATCH, 32, 0, stream>>>(logit, bout, scores, active, lens, seq, tokens, t);
  }
  finalize_k<<<BATCH, 64, 0, stream>>>(seq, scores, lens, out);
}
